// Our_net_80874234184231
// MI455X (gfx1250) — matrix operand order repaired and checked by reading, not compiled, not run on hardware
//
#include <hip/hip_runtime.h>

// ---------------------------------------------------------------------------
// MI455X (gfx1250) implementation of the pansharpening forward pass.
//
// Roofline: ~30 GFLOP vs ~0.5-1 GB feature-map traffic (fits in 192MB L2,
// 23.3 TB/s HBM => ~30-40us memory time). Scalar f32 VALU would be compute
// bound, so all conv3x3 layers and both CSA einsum GEMMs run on
// v_wmma_f32_16x16x32_f16 (f16 in / f32 accumulate), with WMMA A operands
// staged through LDS (pre-padded f16 -> unconditional ds_load_b128 pairs).
// csa_gemm2 stages its B operand with the Tensor Data Mover (builtin if the
// toolchain declares it, else inline asm per the CDNA5 ISA).
// wave32, 8 waves (256 threads) per block.
// ---------------------------------------------------------------------------

typedef __attribute__((ext_vector_type(16))) _Float16 v16h;
typedef __attribute__((ext_vector_type(8)))  _Float16 v8h;
typedef __attribute__((ext_vector_type(8)))  float    v8f;
typedef __attribute__((ext_vector_type(4)))  unsigned int u32x4;
typedef __attribute__((ext_vector_type(8)))  int      i32x8;
typedef __attribute__((ext_vector_type(4)))  int      i32x4;

#define WAVES_PER_BLOCK 8

#define TDM_BUILTIN 0
#define WTC_BUILTIN 0
#if defined(__has_builtin)
#  if __has_builtin(__builtin_amdgcn_tensor_load_to_lds)
#    undef TDM_BUILTIN
#    define TDM_BUILTIN 1
#  endif
#  if __has_builtin(__builtin_amdgcn_s_wait_tensorcnt)
#    undef WTC_BUILTIN
#    define WTC_BUILTIN 1
#  endif
#endif

__device__ __forceinline__ void wait_tensorcnt0() {
#if WTC_BUILTIN
    __builtin_amdgcn_s_wait_tensorcnt(0);
#else
    asm volatile("s_wait_tensorcnt 0x0" ::: "memory");
#endif
}

__device__ __forceinline__ float leaky_f(float x) { return x >= 0.f ? x : 0.01f * x; }
__device__ __forceinline__ int   imin_i(int a, int b) { return a < b ? a : b; }
__device__ __forceinline__ int   imax_i(int a, int b) { return a > b ? a : b; }

__device__ __forceinline__ float keys_cubic(float d) {
    // Keys cubic, a = -0.5 (jax.image.resize 'bicubic')
    float t = fabsf(d);
    if (t <= 1.f) return ((1.5f * t - 2.5f) * t) * t + 1.f;
    if (t <  2.f) return ((-0.5f * t + 2.5f) * t - 4.f) * t + 2.f;
    return 0.f;
}

// concat two v8h into a v16h fragment
#define CAT16(lo, hi) __builtin_shufflevector((lo), (hi), \
        0,1,2,3,4,5,6,7,8,9,10,11,12,13,14,15)

// ---------------------------------------------------------------------------
// Generic 3x3 conv (pad=1, zero or circular), Cin,Cout <= 16, as implicit
// GEMM on v_wmma_f32_16x16x32_f16. One wave computes 16(Cout) x 16(pixel)
// output tile; K = Cin*9 padded to 32-multiples.
// A(16x32 f16): lane l -> M=l%16, half h -> K=(l<16?0:16)+h
// B(32x16 f16): lane l -> N=l%16, half h -> K=(l<16?0:16)+h
// D(16x16 f32): vgpr r, lane l -> M=r+(l<16?0:8), N=l%16
// Weights are staged per-block into LDS, converted to f16 and zero-padded to
// [16][Kpad] so A fragments are unconditional ds_load_b128 pairs.
// ---------------------------------------------------------------------------
__global__ void __launch_bounds__(256)
conv3_wmma(const float* __restrict__ in, const float* __restrict__ w,
           const float* __restrict__ bias, const float* __restrict__ addsrc,
           float* __restrict__ out, _Float16* __restrict__ outh,
           int Nb, int Cin, int Cout, int Hh, int Ww,
           int circular, int leaky, int outCbase, int outCtot)
{
    __shared__ _Float16 wlds[16 * 160];               // up to Kpad=160
    const int lane = threadIdx.x & 31;
    const int wave = threadIdx.x >> 5;
    const int K    = Cin * 9;
    const int nk   = (K + 31) >> 5;
    const int Kpad = nk * 32;

    // cooperative weight staging: f16, zero-padded rows/cols
    for (int m = wave * 2; m < wave * 2 + 2; ++m)
        for (int k = lane; k < Kpad; k += 32)
            wlds[m * Kpad + k] = (m < Cout && k < K) ? (_Float16)w[m * K + k]
                                                     : (_Float16)0.f;
    __syncthreads();

    const int HW = Hh * Ww;
    const int tilesPerImg = (HW + 15) >> 4;
    int tile = blockIdx.x * WAVES_PER_BLOCK + wave;   // wave-uniform
    int n = tile / tilesPerImg;
    int t = tile % tilesPerImg;
    if (n >= Nb) return;                              // uniform per wave

    const int p  = t * 16 + (lane & 15);              // this lane's pixel (B/D col)
    const bool pvalid = p < HW;
    const int pc = imin_i(p, HW - 1);
    const int py = pc / Ww, px = pc % Ww;
    const int kofsA = (lane < 16) ? 0 : 16;
    const int kofsB = (lane < 16) ? 0 : 16;
    const int mA = lane & 15;

    __builtin_prefetch(in + ((size_t)(n * Cin) * Hh + py) * Ww + px, 0, 0);

    v8f acc = {};
    for (int kc = 0; kc < nk; ++kc) {
        // ---- A fragment: two aligned v8h LDS loads (pre-padded) ----
        const v8h* ap = (const v8h*)(wlds + mA * Kpad + kc * 32 + kofsA);
        v16h a = CAT16(ap[0], ap[1]);
        // ---- B fragment: im2col patch, incremental (ci,ky,kx) ----
        const int kB0 = kc * 32 + kofsB;
        int ci  = kB0 / 9;
        int tap = kB0 - ci * 9;
        int ky  = tap / 3 - 1;
        int kx  = tap - (tap / 3) * 3 - 1;
        const float* __restrict__ bp =
            in + ((size_t)n * Cin + (ci < Cin ? ci : Cin - 1)) * HW;
        int kg = kB0;
        v16h b;
#pragma unroll
        for (int h = 0; h < 16; ++h) {
            int yy = py + ky, xx = px + kx;
            bool inb = true;
            if (circular) {
                if (yy < 0) yy += Hh; if (yy >= Hh) yy -= Hh;
                if (xx < 0) xx += Ww; if (xx >= Ww) xx -= Ww;
            } else {
                inb = (yy >= 0) && (yy < Hh) && (xx >= 0) && (xx < Ww);
                yy = imin_i(imax_i(yy, 0), Hh - 1);
                xx = imin_i(imax_i(xx, 0), Ww - 1);
            }
            float bv = bp[yy * Ww + xx];                 // always in-bounds
            b[h] = (_Float16)((inb && kg < K) ? bv : 0.f);
            ++kg;
            if (++kx > 1) { kx = -1;
                if (++ky > 1) { ky = -1; ++ci; if (ci < Cin) bp += HW; } }
        }
        acc = __builtin_amdgcn_wmma_f32_16x16x32_f16(false, a, false, b,
                                                     (short)0, acc, false, false);
    }
#pragma unroll
    for (int r = 0; r < 8; ++r) {
        int m = r + ((lane < 16) ? 0 : 8);
        if (m < Cout && pvalid) {
            float v = acc[r] + bias[m];
            size_t oidx = (((size_t)n * outCtot + outCbase + m) * Hh + py) * Ww + px;
            if (addsrc) v += addsrc[oidx];
            if (leaky)  v = leaky_f(v);
            if (out)  out[oidx]  = v;
            if (outh) outh[oidx] = (_Float16)v;
        }
    }
}

// --------------------------- batchnorm (two pass) ---------------------------
__global__ void __launch_bounds__(256)
bn_stats(const float* __restrict__ x, float* __restrict__ st, int Nb, int C, int HW)
{
    int c = blockIdx.x;
    float s = 0.f, s2 = 0.f;
    for (int i = threadIdx.x; i < Nb * HW; i += 256) {
        int n = i / HW;
        float v = x[((size_t)n * C + c) * HW + (i % HW)];
        s += v; s2 += v * v;
    }
    __shared__ float a1[256], a2[256];
    a1[threadIdx.x] = s; a2[threadIdx.x] = s2;
    __syncthreads();
    for (int o = 128; o > 0; o >>= 1) {
        if (threadIdx.x < o) { a1[threadIdx.x] += a1[threadIdx.x + o];
                               a2[threadIdx.x] += a2[threadIdx.x + o]; }
        __syncthreads();
    }
    if (threadIdx.x == 0) {
        float cnt = (float)(Nb * HW);
        float m = a1[0] / cnt;
        st[c * 2] = m;
        st[c * 2 + 1] = a2[0] / cnt - m * m;
    }
}

__global__ void __launch_bounds__(256)
bn_apply(float* __restrict__ x, const float* __restrict__ st,
         const float* __restrict__ g, const float* __restrict__ b,
         int Nb, int C, int HW, int leaky)
{
    size_t idx = (size_t)blockIdx.x * 256 + threadIdx.x;
    if (idx >= (size_t)Nb * C * HW) return;
    int c = (int)((idx / HW) % C);
    float m = st[c * 2], v = st[c * 2 + 1];
    float y = g[c] * (x[idx] - m) * rsqrtf(v + 1e-5f) + b[c];
    if (leaky) y = leaky_f(y);
    x[idx] = y;
}

// ------------------------- CSA attention pipeline ---------------------------
// kp pack: 100 patches (3x3x10) of k_fea(28x28), L2-normalized by max patch
// norm, written as f16 row-major [n][112][96] (zero padded).
__global__ void __launch_bounds__(128)
csa_kp_pack(const float* __restrict__ kfea, _Float16* __restrict__ kp)
{
    int n = blockIdx.x;
    int l = threadIdx.x;           // 128 threads
    __shared__ float norms[128];
    __shared__ float mx;
    float nrm = 0.f;
    if (l < 100) {
        int i = l / 10, j = l % 10;
        for (int c = 0; c < 10; ++c)
            for (int a = 0; a < 3; ++a)
                for (int b = 0; b < 3; ++b) {
                    int y = 3 * i + a - 1, x = 3 * j + b - 1;
                    float v = (y >= 0 && y < 28 && x >= 0 && x < 28)
                              ? kfea[((n * 10 + c) * 28 + y) * 28 + x] : 0.f;
                    nrm += v * v;
                }
    }
    norms[l] = (l < 100) ? sqrtf(nrm) : 0.f;
    __syncthreads();
    if (threadIdx.x == 0) {
        float m = 0.f;
        for (int q = 0; q < 100; ++q) m = fmaxf(m, norms[q]);
        mx = m;
    }
    __syncthreads();
    float inv = 1.f / mx;
    if (l < 112) {
        for (int k = 0; k < 96; ++k) {
            float v = 0.f;
            if (l < 100 && k < 90) {
                int c = k / 9, tap = k % 9;
                int y = 3 * (l / 10) + tap / 3 - 1, x = 3 * (l % 10) + tap % 3 - 1;
                if (y >= 0 && y < 28 && x >= 0 && x < 28)
                    v = kfea[((n * 10 + c) * 28 + y) * 28 + x] * inv;
            }
            kp[((size_t)n * 112 + l) * 96 + k] = (_Float16)v;
        }
    }
}

#define ATT_P  22500   // 150*150 attention positions
#define ATT_PP 22512   // padded to 16 (1407 tiles)
#define ATT_NT 1407

// GEMM1: w_att[n,l,p] = 10 * sum_k kp[n,l,k] * qpatch[n,k,p]   (M=112,K=96)
// kp (43KB) is staged into LDS once per block (shared by 8 waves x 7 Mtiles).
__global__ void __launch_bounds__(256)
csa_gemm1(const _Float16* __restrict__ qh, const _Float16* __restrict__ kp,
          float* __restrict__ watt)
{
    __shared__ _Float16 klds[2 * 112 * 96];           // 43008 B
    {
        const unsigned* src = (const unsigned*)kp;
        unsigned* dst = (unsigned*)klds;
        for (int i = threadIdx.x; i < 2 * 112 * 96 / 2; i += 256) dst[i] = src[i];
    }
    __syncthreads();

    const int lane = threadIdx.x & 31, wave = threadIdx.x >> 5;
    int id = blockIdx.x * WAVES_PER_BLOCK + wave;
    if (id >= 2 * ATT_NT) return;                 // wave-uniform
    int n = id / ATT_NT, pt = id % ATT_NT;
    int p = pt * 16 + (lane & 15);
    int i = p / 150, j = p % 150;
    const int kofsA = (lane < 16) ? 0 : 16;
    const int kofsB = (lane < 16) ? 0 : 16;
    const int mA = lane & 15;
    v8f acc[7] = {};
    for (int kc = 0; kc < 3; ++kc) {
        // ---- B fragment: q patch gather, incremental (c,ky,kx) ----
        const int kB0 = kc * 32 + kofsB;
        int c   = kB0 / 9;
        int tap = kB0 - c * 9;
        int ky  = tap / 3 - 1;
        int kx  = tap - (tap / 3) * 3 - 1;
        const _Float16* __restrict__ qp =
            qh + ((size_t)n * 10 + (c < 10 ? c : 9)) * (448 * 448);
        const int y0 = 3 * i, x0 = 3 * j;
        int kg = kB0;
        v16h bf;
#pragma unroll
        for (int h = 0; h < 16; ++h) {
            int y = y0 + ky, x = x0 + kx;
            bool inb = (y >= 0) && (y < 448) && (x >= 0) && (x < 448);
            int yc = imin_i(imax_i(y, 0), 447), xc = imin_i(imax_i(x, 0), 447);
            _Float16 qv = qp[yc * 448 + xc];
            bf[h] = (inb && kg < 90) ? qv : (_Float16)0.f;
            ++kg;
            if (++kx > 1) { kx = -1;
                if (++ky > 1) { ky = -1; ++c; if (c < 10) qp += 448 * 448; } }
        }
#pragma unroll
        for (int mt = 0; mt < 7; ++mt) {
            const v8h* ap = (const v8h*)(klds + ((size_t)n * 112 + mt * 16 + mA) * 96
                                         + kc * 32 + kofsA);
            v16h af = CAT16(ap[0], ap[1]);
            acc[mt] = __builtin_amdgcn_wmma_f32_16x16x32_f16(false, af, false, bf,
                                                             (short)0, acc[mt], false, false);
        }
    }
#pragma unroll
    for (int mt = 0; mt < 7; ++mt)
#pragma unroll
        for (int r = 0; r < 8; ++r) {
            int l = mt * 16 + r + ((lane < 16) ? 0 : 8);
            if (l < 100)
                watt[((size_t)n * 100 + l) * ATT_PP + p] = 10.f * acc[mt][r];
        }
}

__global__ void __launch_bounds__(256)
csa_softmax(const float* __restrict__ watt, _Float16* __restrict__ attn)
{
    size_t idx = (size_t)blockIdx.x * 256 + threadIdx.x;
    if (idx >= (size_t)2 * ATT_PP) return;
    int n = (int)(idx / ATT_PP), p = (int)(idx % ATT_PP);
    if (p >= ATT_P) {
        for (int k = 0; k < 128; ++k)
            attn[((size_t)n * 128 + k) * ATT_PP + p] = (_Float16)0.f;
        return;
    }
    float m = -1e30f;
    for (int l = 0; l < 100; ++l)
        m = fmaxf(m, watt[((size_t)n * 100 + l) * ATT_PP + p]);
    float s = 0.f;
    for (int l = 0; l < 100; ++l)
        s += expf(watt[((size_t)n * 100 + l) * ATT_PP + p] - m);
    float inv = 1.f / s;
    for (int l = 0; l < 100; ++l)
        attn[((size_t)n * 128 + l) * ATT_PP + p] =
            (_Float16)(expf(watt[((size_t)n * 100 + l) * ATT_PP + p] - m) * inv);
    for (int k = 100; k < 128; ++k)
        attn[((size_t)n * 128 + k) * ATT_PP + p] = (_Float16)0.f;
}

// vp pack: A2[n][m=c*9+a*3+b][k=l] (36x100 -> padded 48x128), f16
__global__ void __launch_bounds__(256)
csa_vp_pack(const float* __restrict__ vfea, _Float16* __restrict__ vpT)
{
    int n = blockIdx.x;
    for (int idx = threadIdx.x; idx < 48 * 128; idx += 256) {
        int m = idx / 128, k = idx % 128;
        float v = 0.f;
        if (m < 36 && k < 100) {
            int c = m / 9, tap = m % 9;
            int y = 3 * (k / 10) + tap / 3 - 1, x = 3 * (k % 10) + tap % 3 - 1;
            if (y >= 0 && y < 28 && x >= 0 && x < 28)
                v = vfea[((n * 4 + c) * 28 + y) * 28 + x];
        }
        vpT[((size_t)n * 48 + m) * 128 + k] = (_Float16)v;
    }
}

// GEMM2: out450[n,m,p] = sum_l vpT[n,m,l]*attn[n,l,p]; crop to 448 and /6.
// B operand (128 x 16 f16 tile, row stride ATT_PP) is staged into LDS by the
// Tensor Data Mover (one 2D descriptor per wave); A operand (vpT) is staged
// into LDS per block. All WMMA operands are then fed from ds_loads.
__global__ void __launch_bounds__(256)
csa_gemm2(const _Float16* __restrict__ attn, const _Float16* __restrict__ vpT,
          float* __restrict__ prer)
{
    __shared__ _Float16 vplds[2 * 48 * 128];                // 24576 B, LDS off 0
    __shared__ _Float16 btile[WAVES_PER_BLOCK][128 * 16];   // 32768 B, LDS off 24576
    {
        const unsigned* src = (const unsigned*)vpT;
        unsigned* dst = (unsigned*)vplds;
        for (int i = threadIdx.x; i < 2 * 48 * 128 / 2; i += 256) dst[i] = src[i];
    }
    __syncthreads();

    const int lane = threadIdx.x & 31, wave = threadIdx.x >> 5;
    int id = blockIdx.x * WAVES_PER_BLOCK + wave;
    if (id >= 2 * ATT_NT) return;
    int n = id / ATT_NT, pt = id % ATT_NT;
    int p = pt * 16 + (lane & 15);
    const int col = lane & 15;
    const int kofsA = (lane < 16) ? 0 : 16;
    const int kofsB = (lane < 16) ? 0 : 16;
    const int mA = lane & 15;

    {
        // Tensor DMA descriptor (ISA 8.3/8.4): 2D tile, 16 (dim0) x 128 (dim1)
        // f16 elements, tensor_dim0_stride = ATT_PP, into this wave's LDS slab.
        // btile assumed at LDS offset 24576 (declared after vplds).
        unsigned long long ga = (unsigned long long)(uintptr_t)
            (attn + ((size_t)n * 128) * ATT_PP + (size_t)pt * 16);
        u32x4 g0;
        g0.x = 1u;                                            // count=1
        g0.y = 24576u + (unsigned)(wave * (128 * 16 * 2));    // lds_addr (bytes)
        g0.z = (unsigned)(ga & 0xffffffffu);                  // global_addr lo
        g0.w = (unsigned)((ga >> 32) & 0x1ffffffu) | (2u << 30); // hi | type=2
        i32x8 g1;
        g1[0] = 0x00010000;                                   // data_size=2B
        g1[1] = (int)(16u << 16);                             // tensor_dim0=16
        g1[2] = (int)(128u << 16);                            // tensor_dim1=128
        g1[3] = (int)(16u << 16);                             // tile_dim0=16
        g1[4] = 128;                                          // tile_dim1=128
        g1[5] = ATT_PP;                                       // dim0 stride
        g1[6] = 0; g1[7] = 0;
#if TDM_BUILTIN
        i32x4 gz = {0, 0, 0, 0};
#if __clang_major__ >= 23
        i32x8 gz8 = {0, 0, 0, 0, 0, 0, 0, 0};
        __builtin_amdgcn_tensor_load_to_lds(g0, g1, gz, gz, gz8, 0);
#else
        __builtin_amdgcn_tensor_load_to_lds(g0, g1, gz, gz, 0);
#endif
#else
        // Toolchain does not declare the TDM builtin: issue it directly.
        asm volatile("tensor_load_to_lds %0, %1" :: "s"(g0), "s"(g1) : "memory");
#endif
        wait_tensorcnt0();
    }

    v8f acc[3] = {};
    for (int kc = 0; kc < 4; ++kc) {
        v16h bf;
#pragma unroll
        for (int h = 0; h < 16; ++h) {
            int k = kc * 32 + kofsB + h;
            bf[h] = btile[wave][k * 16 + col];
        }
#pragma unroll
        for (int mt = 0; mt < 3; ++mt) {
            const v8h* ap = (const v8h*)(vplds + ((size_t)n * 48 + mt * 16 + mA) * 128
                                         + kc * 32 + kofsA);
            v16h af = CAT16(ap[0], ap[1]);
            acc[mt] = __builtin_amdgcn_wmma_f32_16x16x32_f16(false, af, false, bf,
                                                             (short)0, acc[mt], false, false);
        }
    }
#pragma unroll
    for (int mt = 0; mt < 3; ++mt)
#pragma unroll
        for (int r = 0; r < 8; ++r) {
            int m = mt * 16 + r + ((lane < 16) ? 0 : 8);
            if (m < 36 && p < ATT_P) {
                int c = m / 9, tap = m % 9;
                int y = 3 * (p / 150) + tap / 3;   // 450-grid coords
                int x = 3 * (p % 150) + tap % 3;
                if (y >= 1 && y <= 448 && x >= 1 && x <= 448)
                    prer[(((size_t)n * 4 + c) * 448 + (y - 1)) * 448 + (x - 1)]
                        = acc[mt][r] * (1.f / 6.f);
            }
        }
}

// ----------------------------- CSA-spe (tiny) -------------------------------
__global__ void spe_scalars(const float* __restrict__ pan2, const float* __restrict__ ms2,
                            const float* __restrict__ qw, const float* __restrict__ qb,
                            const float* __restrict__ kw, const float* __restrict__ kb,
                            float* __restrict__ att)
{
    __shared__ float sc[2][4];
    int tid = threadIdx.x;
    if (tid < 8) {
        int n = tid / 4, c = tid % 4;
        // q: conv(pan2 28x28, stride 256, pad 1) -> single value per n
        float qv = qb[0];
        for (int a = 0; a < 3; ++a)
            for (int b = 0; b < 3; ++b) {
                int y = a - 1, x = b - 1;
                if (y >= 0 && y < 28 && x >= 0 && x < 28)
                    qv += qw[a * 3 + b] * pan2[(n * 28 + y) * 28 + x];
            }
        qv = leaky_f(qv);
        float qn = qv / fabsf(qv);
        // k: conv(ms2 2x2, stride 16, pad 1) -> single value per (n,c)
        float kv = kb[c];
        for (int ci = 0; ci < 4; ++ci)
            for (int a = 0; a < 3; ++a)
                for (int b = 0; b < 3; ++b) {
                    int y = a - 1, x = b - 1;
                    if (y >= 0 && y < 2 && x >= 0 && x < 2)
                        kv += kw[((c * 4 + ci) * 3 + a) * 3 + b]
                              * ms2[((n * 4 + ci) * 2 + y) * 2 + x];
                }
        kv = leaky_f(kv);
        sc[n][c] = kv * qn * 10.f;
    }
    __syncthreads();
    if (tid < 8) {
        int n = tid / 4, c = tid % 4;
        float m = fmaxf(fmaxf(sc[n][0], sc[n][1]), fmaxf(sc[n][2], sc[n][3]));
        float s = expf(sc[n][0] - m) + expf(sc[n][1] - m)
                + expf(sc[n][2] - m) + expf(sc[n][3] - m);
        att[n * 4 + c] = expf(sc[n][c] - m) / s;
    }
}

__global__ void __launch_bounds__(256)
spe_v(const float* __restrict__ pan, const float* __restrict__ vw,
      const float* __restrict__ vb, const float* __restrict__ att,
      float* __restrict__ outp)
{
    size_t idx = (size_t)blockIdx.x * 256 + threadIdx.x;
    if (idx >= (size_t)2 * 448 * 448) return;
    int n = (int)(idx / (448 * 448));
    int y = (int)((idx / 448) % 448), x = (int)(idx % 448);
    float s = vb[0];
    for (int a = 0; a < 3; ++a)
        for (int b = 0; b < 3; ++b) {
            int yy = y + a - 1, xx = x + b - 1;
            if (yy >= 0 && yy < 448 && xx >= 0 && xx < 448)
                s += vw[a * 3 + b] * pan[((size_t)n * 448 + yy) * 448 + xx];
        }
    s = leaky_f(s);
    for (int c = 0; c < 4; ++c)
        outp[(((size_t)n * 4 + c) * 448 + y) * 448 + x] = s * att[n * 4 + c] * (1.f / 6.f);
}

// --------------------------- pointwise / blur ops ---------------------------
__global__ void __launch_bounds__(256)
blur5_grouped(const float* __restrict__ in, const float* __restrict__ w,
              const float* __restrict__ b, float* __restrict__ out,
              int Nb, int C, int H, int W)
{
    size_t idx = (size_t)blockIdx.x * 256 + threadIdx.x;
    if (idx >= (size_t)Nb * C * H * W) return;
    int x = (int)(idx % W), y = (int)((idx / W) % H);
    int c = (int)((idx / ((size_t)H * W)) % C), n = (int)(idx / ((size_t)C * H * W));
    float s = b[c];
#pragma unroll 1
    for (int ky = 0; ky < 5; ++ky) {
        int yy = (y + ky - 2 + H) % H;
        for (int kx = 0; kx < 5; ++kx) {
            int xx = (x + kx - 2 + W) % W;
            s += w[c * 25 + ky * 5 + kx] * in[(((size_t)n * C + c) * H + yy) * W + xx];
        }
    }
    out[idx] = s;
}

__global__ void __launch_bounds__(256)
down16(const float* __restrict__ in, float* __restrict__ out, int Nb, int C, int H, int W)
{
    int HO = (H + 15) / 16, WO = (W + 15) / 16;
    size_t idx = (size_t)blockIdx.x * 256 + threadIdx.x;
    if (idx >= (size_t)Nb * C * HO * WO) return;
    int x = (int)(idx % WO), y = (int)((idx / WO) % HO);
    int c = (int)((idx / ((size_t)HO * WO)) % C), n = (int)(idx / ((size_t)C * HO * WO));
    out[idx] = in[(((size_t)n * C + c) * H + y * 16) * W + x * 16];
}

__global__ void __launch_bounds__(256)
mean_c(const float* __restrict__ in, float* __restrict__ out, int Nb, int C, int HW)
{
    size_t idx = (size_t)blockIdx.x * 256 + threadIdx.x;
    if (idx >= (size_t)Nb * HW) return;
    int n = (int)(idx / HW), p = (int)(idx % HW);
    float s = 0.f;
    for (int c = 0; c < 4; ++c) s += in[((size_t)n * C + c) * HW + p];
    out[idx] = s / (float)C;
}

__global__ void __launch_bounds__(256)
conv11_circ(const float* __restrict__ in, const float* __restrict__ w,
            const float* __restrict__ b, float* __restrict__ out)
{
    size_t idx = (size_t)blockIdx.x * 256 + threadIdx.x;
    if (idx >= (size_t)2 * 8 * 448 * 448) return;
    int x = (int)(idx % 448), y = (int)((idx / 448) % 448);
    int co = (int)((idx / (448 * 448)) % 8), n = (int)(idx / (8ull * 448 * 448));
    float s = b[co];
#pragma unroll 1
    for (int ci = 0; ci < 8; ++ci)
#pragma unroll 1
        for (int ky = 0; ky < 11; ++ky) {
            int yy = (y + ky - 5 + 448) % 448;
            const float* row = in + (((size_t)n * 8 + ci) * 448 + yy) * 448;
            const float* wr  = w + ((co * 8 + ci) * 11 + ky) * 11;
#pragma unroll 1
            for (int kx = 0; kx < 11; ++kx) {
                int xx = (x + kx - 5 + 448) % 448;
                s += wr[kx] * row[xx];
            }
        }
    out[idx] = s;
}

// dd2 (1x1, 8->4) + bicubic upsample of ms (28->448, Keys a=-0.5,
// edge weights renormalized like jax compute_weight_mat) + add.
__global__ void __launch_bounds__(256)
dd2_up_add(const float* __restrict__ dd, const float* __restrict__ w2,
           const float* __restrict__ b2, const float* __restrict__ ms,
           float* __restrict__ out)
{
    size_t idx = (size_t)blockIdx.x * 256 + threadIdx.x;
    if (idx >= (size_t)2 * 4 * 448 * 448) return;
    int x = (int)(idx % 448), y = (int)((idx / 448) % 448);
    int c = (int)((idx / (448 * 448)) % 4), n = (int)(idx / (4ull * 448 * 448));
    float s = b2[c];
    for (int ci = 0; ci < 8; ++ci)
        s += w2[c * 8 + ci] * dd[(((size_t)n * 8 + ci) * 448 + y) * 448 + x];
    // bicubic
    float sy = (y + 0.5f) * (1.f / 16.f) - 0.5f;
    float sx = (x + 0.5f) * (1.f / 16.f) - 0.5f;
    int by = (int)floorf(sy), bx = (int)floorf(sx);
    float wy[4], wx[4]; int iy[4], ix[4]; float wys = 0.f, wxs = 0.f;
#pragma unroll
    for (int t = 0; t < 4; ++t) {
        int pos = by - 1 + t;
        bool ok = pos >= 0 && pos < 28;
        wy[t] = ok ? keys_cubic(sy - (float)pos) : 0.f;
        wys += wy[t]; iy[t] = ok ? pos : 0;
        pos = bx - 1 + t;
        ok = pos >= 0 && pos < 28;
        wx[t] = ok ? keys_cubic(sx - (float)pos) : 0.f;
        wxs += wx[t]; ix[t] = ok ? pos : 0;
    }
    float up = 0.f;
#pragma unroll
    for (int ty = 0; ty < 4; ++ty)
#pragma unroll
        for (int tx = 0; tx < 4; ++tx)
            up += wy[ty] * wx[tx] * ms[((n * 4 + c) * 28 + iy[ty]) * 28 + ix[tx]];
    up /= (wys * wxs);
    out[idx] = s + up;
}

// ----------------------------- host-side driver -----------------------------
static inline int ceil_div(long long a, long long b) { return (int)((a + b - 1) / b); }

static void conv3(const float* in, const float* w, const float* bias, const float* addsrc,
                  float* out, _Float16* outh, int Nb, int Cin, int Cout, int H, int W,
                  int circ, int leaky, int outCbase, int outCtot, hipStream_t s)
{
    int tiles = Nb * ((H * W + 15) / 16);
    conv3_wmma<<<ceil_div(tiles, WAVES_PER_BLOCK), 256, 0, s>>>(
        in, w, bias, addsrc, out, outh, Nb, Cin, Cout, H, W, circ, leaky, outCbase, outCtot);
}

static void run_encoder(const float* x, int Cin, int H, int W,
                        const float* const* P, float* A, float* B,
                        float* stats, hipStream_t s)
{
    int HW = H * W;
    int gel = ceil_div(2LL * 10 * HW, 256);
    conv3(x, P[0], P[1], nullptr, A, nullptr, 2, Cin, 10, H, W, 1, 0, 0, 10, s);
    bn_stats<<<10, 256, 0, s>>>(A, stats, 2, 10, HW);
    bn_apply<<<gel, 256, 0, s>>>(A, stats, P[2], P[3], 2, 10, HW, 1);
    conv3(A, P[4], P[5], nullptr, B, nullptr, 2, 10, 10, H, W, 1, 0, 0, 10, s);
    bn_stats<<<10, 256, 0, s>>>(B, stats, 2, 10, HW);
    bn_apply<<<gel, 256, 0, s>>>(B, stats, P[6], P[7], 2, 10, HW, 0);
    for (int r = 0; r < 5; ++r) {
        conv3(B, P[8 + 4 * r], P[9 + 4 * r], nullptr, A, nullptr, 2, 10, 10, H, W, 1, 1, 0, 10, s);
        conv3(A, P[10 + 4 * r], P[11 + 4 * r], B, B, nullptr, 2, 10, 10, H, W, 1, 0, 0, 10, s);
    }
}

extern "C" void kernel_launch(void* const* d_in, const int* in_sizes, int n_in,
                              void* d_out, int out_size, void* d_ws, size_t ws_size,
                              hipStream_t stream)
{
    // ---- inputs (setup_inputs dict order, nested dicts/lists flattened) ----
    int t = 0;
    const float* ms  = (const float*)d_in[t++];   // (2,4,28,28)
    const float* pan = (const float*)d_in[t++];   // (2,1,448,448)
    const float* blur_ms_w  = (const float*)d_in[t++];
    const float* blur_ms_b  = (const float*)d_in[t++];
    const float* blur_pan_w = (const float*)d_in[t++];
    const float* blur_pan_b = (const float*)d_in[t++];
    const float* encms[28];  for (int i = 0; i < 28; ++i) encms[i]  = (const float*)d_in[t++];
    const float* encpan[28]; for (int i = 0; i < 28; ++i) encpan[i] = (const float*)d_in[t++];
    const float* csa_q_w = (const float*)d_in[t++]; const float* csa_q_b = (const float*)d_in[t++];
    const float* csa_k_w = (const float*)d_in[t++]; const float* csa_k_b = (const float*)d_in[t++];
    const float* csa_v_w = (const float*)d_in[t++]; const float* csa_v_b = (const float*)d_in[t++];
    const float* csa_r_w = (const float*)d_in[t++]; const float* csa_r_b = (const float*)d_in[t++];
    const float* spe_v_w = (const float*)d_in[t++]; const float* spe_v_b = (const float*)d_in[t++];
    const float* spe_q_w = (const float*)d_in[t++]; const float* spe_q_b = (const float*)d_in[t++];
    const float* spe_k_w = (const float*)d_in[t++]; const float* spe_k_b = (const float*)d_in[t++];
    const float* spe_r_w = (const float*)d_in[t++]; const float* spe_r_b = (const float*)d_in[t++];
    const float* dd1_w = (const float*)d_in[t++]; const float* dd1_b = (const float*)d_in[t++];
    const float* dd2_w = (const float*)d_in[t++]; const float* dd2_b = (const float*)d_in[t++];

    // ---- output offsets (floats, return-tuple order) ----
    float* out = (float*)d_out;
    float* o_fuse0   = out;                  // (2,4,448,448)
    float* o_ms2     = out + 1605632;        // (2,4,2,2)
    float* o_pan2    = out + 1605664;        // (2,1,28,28)
    float* o_srf     = out + 1607232;        // (2,1,28,28)
    float* o_f14     = out + 1608800;        // (2,4,28,28)
    float* o_psrf    = out + 1615072;        // (2,1,448,448)
    float* o_blurms  = out + 2016480;        // (2,4,28,28)
    float* o_blurpan = out + 2022752;        // (2,1,448,448)
    float* o_blurf0  = out + 2424160;        // (2,4,448,448)

    // ---- workspace carve ----
    char* wp = (char*)d_ws;
    auto carve = [&](size_t bytes) { char* p = wp; wp += (bytes + 255) & ~(size_t)255; return p; };
    float*    ws_panf  = (float*)   carve(4014080ull * 4);        // pan_f / encoder B
    float*    ws_t10   = (float*)   carve(4014080ull * 4);        // encoder A
    float*    ws_msf   = (float*)   carve(15680ull * 4);          // ms_f
    float*    ws_mst   = (float*)   carve(15680ull * 4);          // ms enc temp / k_fea
    _Float16* ws_qh    = (_Float16*)carve(4014080ull * 2);        // q (f16)
    _Float16* ws_kp    = (_Float16*)carve(2ull * 112 * 96 * 2);
    float*    ws_watt  = (float*)   carve(2ull * 100 * ATT_PP * 4);
    _Float16* ws_attn  = (_Float16*)carve(2ull * 128 * ATT_PP * 2);
    float*    ws_vfea  = (float*)   carve(6272ull * 4);
    _Float16* ws_vpT   = (_Float16*)carve(2ull * 48 * 128 * 2);
    float*    ws_prer  = (float*)   carve(2ull * 4 * 448 * 448 * 4);
    float*    ws_ddin  = (float*)   carve(2ull * 8 * 448 * 448 * 4);
    float*    ws_ddout = (float*)   carve(2ull * 8 * 448 * 448 * 4);
    float*    ws_stats = (float*)   carve(64 * 4);
    float*    ws_spea  = (float*)   carve(8 * 4);

    // 1) blurs + downsamples + channel mean
    blur5_grouped<<<ceil_div(2LL*4*28*28, 256), 256, 0, stream>>>(ms, blur_ms_w, blur_ms_b, o_blurms, 2, 4, 28, 28);
    down16<<<1, 256, 0, stream>>>(o_blurms, o_ms2, 2, 4, 28, 28);
    blur5_grouped<<<ceil_div(2LL*448*448, 256), 256, 0, stream>>>(pan, blur_pan_w, blur_pan_b, o_blurpan, 2, 1, 448, 448);
    down16<<<ceil_div(2LL*28*28, 256), 256, 0, stream>>>(o_blurpan, o_pan2, 2, 1, 448, 448);
    mean_c<<<ceil_div(2LL*28*28, 256), 256, 0, stream>>>(ms, o_srf, 2, 4, 28 * 28);

    // 2) encoders (WMMA convs + BN)
    run_encoder(ms, 4, 28, 28, encms, ws_mst, ws_msf, ws_stats, stream);
    run_encoder(pan, 1, 448, 448, encpan, ws_t10, ws_panf, ws_stats, stream);

    // 3) CSA spatial attention
    conv3(ws_panf, csa_q_w, csa_q_b, nullptr, nullptr, ws_qh, 2, 10, 10, 448, 448, 0, 1, 0, 10, stream);
    conv3(ws_msf, csa_k_w, csa_k_b, nullptr, ws_mst, nullptr, 2, 10, 10, 28, 28, 0, 1, 0, 10, stream);
    csa_kp_pack<<<2, 128, 0, stream>>>(ws_mst, ws_kp);
    csa_gemm1<<<ceil_div(2LL * ATT_NT, WAVES_PER_BLOCK), 256, 0, stream>>>(ws_qh, ws_kp, ws_watt);
    csa_softmax<<<ceil_div(2LL * ATT_PP, 256), 256, 0, stream>>>(ws_watt, ws_attn);
    conv3(ms, csa_v_w, csa_v_b, nullptr, ws_vfea, nullptr, 2, 4, 4, 28, 28, 0, 1, 0, 4, stream);
    csa_vp_pack<<<2, 256, 0, stream>>>(ws_vfea, ws_vpT);
    csa_gemm2<<<ceil_div(2LL * ATT_NT, WAVES_PER_BLOCK), 256, 0, stream>>>(ws_attn, ws_vpT, ws_prer);
    conv3(ws_prer, csa_r_w, csa_r_b, nullptr, ws_ddin, nullptr, 2, 4, 4, 448, 448, 0, 1, 0, 8, stream);

    // 4) CSA-spe spectral attention
    spe_scalars<<<1, 32, 0, stream>>>(o_pan2, o_ms2, spe_q_w, spe_q_b, spe_k_w, spe_k_b, ws_spea);
    spe_v<<<ceil_div(2LL*448*448, 256), 256, 0, stream>>>(pan, spe_v_w, spe_v_b, ws_spea, ws_prer);
    conv3(ws_prer, spe_r_w, spe_r_b, nullptr, ws_ddin, nullptr, 2, 4, 4, 448, 448, 0, 1, 4, 8, stream);

    // 5) dd1 (11x11 circular) -> dd2(1x1) + bicubic up + add -> ms_fuse0
    conv11_circ<<<ceil_div(2LL*8*448*448, 256), 256, 0, stream>>>(ws_ddin, dd1_w, dd1_b, ws_ddout);
    dd2_up_add<<<ceil_div(2LL*4*448*448, 256), 256, 0, stream>>>(ws_ddout, dd2_w, dd2_b, ms, o_fuse0);

    // 6) final blur / downsample / mean
    blur5_grouped<<<ceil_div(2LL*4*448*448, 256), 256, 0, stream>>>(o_fuse0, blur_ms_w, blur_ms_b, o_blurf0, 2, 4, 448, 448);
    down16<<<ceil_div(2LL*4*28*28, 256), 256, 0, stream>>>(o_blurf0, o_f14, 2, 4, 448, 448);
    mean_c<<<ceil_div(2LL*448*448, 256), 256, 0, stream>>>(o_fuse0, o_psrf, 2, 4, 448 * 448);
}
